// MFA_41111426957320
// MI455X (gfx1250) — compile-verified, matrix operand order checked by reading, not run on hardware
//
#include <hip/hip_runtime.h>
#include <hip/hip_bf16.h>
#include <math.h>

// MFA per-component log-likelihood for MI455X (gfx1250, wave32, WMMA).
// K=64 components, d=4096, l=64 latent, N=2048 samples. Output (N,K) f32.

typedef __attribute__((ext_vector_type(16))) __bf16 v16bf;
typedef __attribute__((ext_vector_type(8)))  float  v8f;

#define KC 64
#define DDIM 4096
#define LLAT 64
#define NSAMP 2048
#define LOG2PI 1.8378770664093453f

// ---------------------------------------------------------------------------
// bf16 split (hi + lo ~= f32) for 3-term WMMA emulation of f32 GEMM
// ---------------------------------------------------------------------------
__device__ __forceinline__ void split_bf16(float x, __bf16& hi, __bf16& lo) {
  hi = (__bf16)x;
  lo = (__bf16)(x - (float)hi);
}

// Gather one 16x32 bf16 WMMA operand from an LDS row that is contiguous in K.
// Per ISA 7.12.2 (16-bit A-matrix 16x32): lanes 0-15 hold K=0..7 (VGPR0-3) and
// K=16..23 (VGPR4-7); lanes 16-31 hold K=8..15 and K=24..31.  The B operand
// (32x16) has the mirrored layout, so the same gather works when `row` is the
// column-major (contiguous-in-K) column of B.
__device__ __forceinline__ v16bf gather16(const __bf16* row, int half) {
  v16bf v;
  const __bf16* p = row + (half << 3);
#pragma unroll
  for (int e = 0; e < 8; ++e) {
    v[e]     = p[e];        // K = 8*half + e
    v[e + 8] = p[e + 16];   // K = 16 + 8*half + e
  }
  return v;
}

// ---------------------------------------------------------------------------
// Kernel 1: iD = exp(-log_D); per-k sum of log_D
// ---------------------------------------------------------------------------
__global__ void __launch_bounds__(256)
prep_kernel(const float* __restrict__ log_D, float* __restrict__ iD,
            float* __restrict__ sumlogD) {
  const int k = blockIdx.x, t = threadIdx.x;
  float s = 0.f;
  for (int d = t; d < DDIM; d += 256) {
    float ld = log_D[k * DDIM + d];
    iD[k * DDIM + d] = expf(-ld);
    s += ld;
  }
  __shared__ float red[256];
  red[t] = s;
  __syncthreads();
  for (int o = 128; o > 0; o >>= 1) {
    if (t < o) red[t] += red[t + o];
    __syncthreads();
  }
  if (t == 0) sumlogD[k] = red[0];
}

// ---------------------------------------------------------------------------
// Kernel 2: PI = softmax(PI_logits), K=64, one block of 64 threads
// ---------------------------------------------------------------------------
__global__ void __launch_bounds__(64)
softmax_kernel(const float* __restrict__ logits, float* __restrict__ PI) {
  __shared__ float sh[64];
  const int t = threadIdx.x;
  float v = logits[t];
  sh[t] = v;
  __syncthreads();
  float m = sh[0];
  for (int i = 1; i < 64; ++i) m = fmaxf(m, sh[i]);
  __syncthreads();
  float e = expf(v - m);
  sh[t] = e;
  __syncthreads();
  float s = 0.f;
  for (int i = 0; i < 64; ++i) s += sh[i];
  PI[t] = e / s;
}

// ---------------------------------------------------------------------------
// Kernel 3: c[k,l] = sum_d iD[k,d]*MU[k,d]*A[k,d,l]   (t offset)
// ---------------------------------------------------------------------------
__global__ void __launch_bounds__(64)
toffset_kernel(const float* __restrict__ MU, const float* __restrict__ A,
               const float* __restrict__ iD, float* __restrict__ coff) {
  const int k = blockIdx.x, l = threadIdx.x;
  const float* Ak = A + (size_t)k * DDIM * LLAT;
  float acc = 0.f;
  for (int d = 0; d < DDIM; ++d) {
    float w = iD[k * DDIM + d] * MU[k * DDIM + d];   // broadcast loads
    acc += w * Ak[(size_t)d * LLAT + l];             // coalesced in l
  }
  coff[k * LLAT + l] = acc;
}

// ---------------------------------------------------------------------------
// Kernel 4: L[k] = I + (iD o A)^T (iD o A) via bf16x3 WMMA.
// One block per k; 16 waves each own one 16x16 tile of the 64x64 output.
// ---------------------------------------------------------------------------
__global__ void __launch_bounds__(512)
build_L_kernel(const float* __restrict__ A, const float* __restrict__ iD,
               float* __restrict__ Lmat) {
  const int k    = blockIdx.x;
  const int tid  = threadIdx.x;
  const int wave = tid >> 5;
  const int lane = tid & 31;
  const int half = lane >> 4;
  const int mrow = lane & 15;
  const int li = wave >> 2, lj = wave & 3;

  __shared__ __bf16 ATh[64][40];   // AT[l][dd] = iD[d0+dd]*A[d0+dd][l], hi part
  __shared__ __bf16 ATl[64][40];   // lo part

  const float* Ak  = A + (size_t)k * DDIM * LLAT;
  const float* iDk = iD + (size_t)k * DDIM;

  v8f acc = {};
  for (int d0 = 0; d0 < DDIM; d0 += 32) {
    __syncthreads();
    {  // stage: thread -> dd = tid/16, 4 consecutive l values (coalesced)
      const int dd = tid >> 4;
      const int l4 = (tid & 15) << 2;
      const float w = iDk[d0 + dd];
      const float* src = Ak + (size_t)(d0 + dd) * LLAT + l4;
#pragma unroll
      for (int j = 0; j < 4; ++j) {
        __bf16 hi, lo;
        split_bf16(w * src[j], hi, lo);
        ATh[l4 + j][dd] = hi;
        ATl[l4 + j][dd] = lo;
      }
    }
    __syncthreads();
    v16bf ah = gather16(&ATh[li * 16 + mrow][0], half);
    v16bf al = gather16(&ATl[li * 16 + mrow][0], half);
    v16bf bh = gather16(&ATh[lj * 16 + mrow][0], half);
    v16bf bl = gather16(&ATl[lj * 16 + mrow][0], half);
    acc = __builtin_amdgcn_wmma_f32_16x16x32_bf16(false, al, false, bh, (short)0, acc, false, false);
    acc = __builtin_amdgcn_wmma_f32_16x16x32_bf16(false, ah, false, bl, (short)0, acc, false, false);
    acc = __builtin_amdgcn_wmma_f32_16x16x32_bf16(false, ah, false, bh, (short)0, acc, false, false);
  }

  float* Lk = Lmat + (size_t)k * LLAT * LLAT;
#pragma unroll
  for (int r = 0; r < 8; ++r) {  // C/D layout: VGPR r -> M = r + 8*half, N = mrow
    const int gr = li * 16 + r + half * 8;
    const int gc = lj * 16 + mrow;
    Lk[gr * LLAT + gc] = acc[r] + ((gr == gc) ? 1.0f : 0.0f);
  }
}

// ---------------------------------------------------------------------------
// Kernel 5: per-k Gauss-Jordan inversion of SPD 64x64 L in LDS; also logdet.
// One block (64 threads, 1 row/thread) per k.
// ---------------------------------------------------------------------------
__global__ void __launch_bounds__(64)
ginv_kernel(const float* __restrict__ Lmat, const float* __restrict__ sumlogD,
            float* __restrict__ iL, float* __restrict__ logdet) {
  const int k = blockIdx.x, i = threadIdx.x;
  __shared__ float M[64][65];
  for (int j = 0; j < 64; ++j) M[i][j] = Lmat[(size_t)k * LLAT * LLAT + i * LLAT + j];
  __syncthreads();
  float ld = 0.f;
  for (int p = 0; p < 64; ++p) {
    const float piv = M[p][p];          // all threads read pivot
    __syncthreads();
    if (i == p) {
      const float inv = 1.0f / piv;
      for (int j = 0; j < 64; ++j) M[p][j] *= inv;
      M[p][p] = inv;
    }
    __syncthreads();
    if (i != p) {
      const float f = M[i][p];
      for (int j = 0; j < 64; ++j) M[i][j] -= f * M[p][j];
      M[i][p] = -f * M[p][p];
    }
    ld += logf(piv);                    // SPD: pivots positive
    __syncthreads();
  }
  for (int j = 0; j < 64; ++j) iL[(size_t)k * LLAT * LLAT + i * LLAT + j] = M[i][j];
  if (i == 0) logdet[k] = ld + sumlogD[k];
}

// ---------------------------------------------------------------------------
// Kernel 6 (main): fused per (k, 64-row n-block):
//   t = x @ (iD o A)  (bf16x3 WMMA over d=4096)  minus offset c[k,:]
//   q_diag = sum_d iD*(x-MU)^2   (f32 VALU riding the x staging)
//   q_low  = t iL t^T row-dots   (f32 VALU on LDS tile)
//   out[n,k] = PI[k] - 0.5*(d*log2pi + logdet[k] + q_diag - q_low)
// ---------------------------------------------------------------------------
__global__ void __launch_bounds__(512)
mfa_main_kernel(const float* __restrict__ x, const float* __restrict__ MU,
                const float* __restrict__ A, const float* __restrict__ iD,
                const float* __restrict__ iL, const float* __restrict__ coff,
                const float* __restrict__ PI, const float* __restrict__ logdet,
                float* __restrict__ out) {
  const int k   = blockIdx.x;
  const int n0  = blockIdx.y * 64;
  const int tid = threadIdx.x;
  const int wave = tid >> 5, lane = tid & 31;
  const int half = lane >> 4, mrow = lane & 15;
  const int ni = wave >> 2, lj = wave & 3;   // n-subtile, l-subtile

  __shared__ __bf16 Xh[64][40], Xl[64][40];  // x rows, contiguous in d
  __shared__ __bf16 Bh[64][40], Bl[64][40];  // (iD o A) columns, contiguous in d
  __shared__ float  t_lds[64][65];
  __shared__ float  iL_lds[64][65];
  __shared__ float  c_lds[64];
  __shared__ float  qd8[64][8];              // staged deterministic reductions
  __shared__ float  ql8[64][8];

  for (int idx = tid; idx < 64 * 64; idx += 512)
    iL_lds[idx >> 6][idx & 63] = iL[(size_t)k * LLAT * LLAT + idx];
  if (tid < 64) c_lds[tid] = coff[k * LLAT + tid];

  const float* Ak  = A  + (size_t)k * DDIM * LLAT;
  const float* iDk = iD + (size_t)k * DDIM;
  const float* MUk = MU + (size_t)k * DDIM;

  // q_diag staging assignment is loop-invariant: thread owns row xn, 4 d-cols
  const int xn  = tid >> 3;
  const int xd4 = (tid & 7) << 2;
  float qpart = 0.f;

  v8f acc = {};
  for (int d0 = 0; d0 < DDIM; d0 += 32) {
    __syncthreads();
    {  // stage X (and accumulate diagonal quadratic term)
      const float* xs = x + (size_t)(n0 + xn) * DDIM + d0 + xd4;
#pragma unroll
      for (int j = 0; j < 4; ++j) {
        const int dd = xd4 + j;
        const float xv = xs[j];
        const float diff = xv - MUk[d0 + dd];
        qpart += iDk[d0 + dd] * diff * diff;
        __bf16 hi, lo;
        split_bf16(xv, hi, lo);
        Xh[xn][dd] = hi;
        Xl[xn][dd] = lo;
      }
    }
    {  // stage B = iD o A
      const int dd = tid >> 4;
      const int l4 = (tid & 15) << 2;
      const float w = iDk[d0 + dd];
      const float* src = Ak + (size_t)(d0 + dd) * LLAT + l4;
#pragma unroll
      for (int j = 0; j < 4; ++j) {
        __bf16 hi, lo;
        split_bf16(w * src[j], hi, lo);
        Bh[l4 + j][dd] = hi;
        Bl[l4 + j][dd] = lo;
      }
    }
    __syncthreads();
    v16bf ah = gather16(&Xh[ni * 16 + mrow][0], half);
    v16bf al = gather16(&Xl[ni * 16 + mrow][0], half);
    v16bf bh = gather16(&Bh[lj * 16 + mrow][0], half);
    v16bf bl = gather16(&Bl[lj * 16 + mrow][0], half);
    acc = __builtin_amdgcn_wmma_f32_16x16x32_bf16(false, al, false, bh, (short)0, acc, false, false);
    acc = __builtin_amdgcn_wmma_f32_16x16x32_bf16(false, ah, false, bl, (short)0, acc, false, false);
    acc = __builtin_amdgcn_wmma_f32_16x16x32_bf16(false, ah, false, bh, (short)0, acc, false, false);
  }

  qd8[xn][tid & 7] = qpart;
  {  // t tile (minus offset) -> LDS
    const int col = lj * 16 + mrow;
    const float cv = c_lds[col];
#pragma unroll
    for (int r = 0; r < 8; ++r)
      t_lds[ni * 16 + r + half * 8][col] = acc[r] - cv;
  }
  __syncthreads();

  {  // q_low: 8 threads per row, 8 m-columns each
    const int n  = tid >> 3;
    const int m0 = (tid & 7) << 3;
    float part = 0.f;
#pragma unroll
    for (int mm = 0; mm < 8; ++mm) {
      const int m = m0 + mm;
      float u = 0.f;
      for (int l = 0; l < 64; ++l) u += t_lds[n][l] * iL_lds[l][m];
      part += u * t_lds[n][m];
    }
    ql8[n][tid & 7] = part;
  }
  __syncthreads();

  if (tid < 64) {
    const int n = tid;
    float qdv = 0.f, qlv = 0.f;
#pragma unroll
    for (int j = 0; j < 8; ++j) { qdv += qd8[n][j]; qlv += ql8[n][j]; }
    const float md = qdv - qlv;
    const float lp = -0.5f * ((float)DDIM * LOG2PI + logdet[k] + md);
    out[(size_t)(n0 + n) * KC + k] = PI[k] + lp;
  }
}

// ---------------------------------------------------------------------------
// Host launcher
// ---------------------------------------------------------------------------
extern "C" void kernel_launch(void* const* d_in, const int* in_sizes, int n_in,
                              void* d_out, int out_size, void* d_ws, size_t ws_size,
                              hipStream_t stream) {
  (void)in_sizes; (void)n_in; (void)out_size; (void)ws_size;
  const float* x     = (const float*)d_in[0];
  const float* MU    = (const float*)d_in[1];
  const float* A     = (const float*)d_in[2];
  const float* log_D = (const float*)d_in[3];
  const float* PIlog = (const float*)d_in[4];
  float* out = (float*)d_out;

  // Workspace layout (floats): ~3.2 MB total
  float* ws    = (float*)d_ws;
  float* iD    = ws;                       // K*D
  float* Lm    = iD   + (size_t)KC * DDIM; // K*64*64
  float* iLm   = Lm   + (size_t)KC * LLAT * LLAT;
  float* coff  = iLm  + (size_t)KC * LLAT * LLAT;  // K*64
  float* slog  = coff + (size_t)KC * LLAT;         // K
  float* PIw   = slog + KC;                        // K
  float* ldet  = PIw  + KC;                        // K

  prep_kernel   <<<dim3(KC), dim3(256), 0, stream>>>(log_D, iD, slog);
  softmax_kernel<<<dim3(1),  dim3(64),  0, stream>>>(PIlog, PIw);
  toffset_kernel<<<dim3(KC), dim3(64),  0, stream>>>(MU, A, iD, coff);
  build_L_kernel<<<dim3(KC), dim3(512), 0, stream>>>(A, iD, Lm);
  ginv_kernel   <<<dim3(KC), dim3(64),  0, stream>>>(Lm, slog, iLm, ldet);
  mfa_main_kernel<<<dim3(KC, NSAMP / 64), dim3(512), 0, stream>>>(
      x, MU, A, iD, iLm, coff, PIw, ldet, out);
}
